// CaptionNet_23467701305971
// MI455X (gfx1250) — matrix operand order repaired and checked by reading, not compiled, not run on hardware
//
#include <hip/hip_runtime.h>

typedef __attribute__((ext_vector_type(16))) _Float16 v16h;
typedef __attribute__((ext_vector_type(8)))  float    v8f;

union H16 { v16h v; _Float16 h[16]; uint4 u[2]; };
union F8  { v8f  v; float f[8]; };

// Problem dims
#define B_  16
#define N_  5
#define T_  10
#define H_  512
#define E_  250
#define EP_ 256
#define V_  30000
#define F_  2048
#define G4_ 2048   // 4*H
#define H2_ 1024   // 2*H

__device__ __forceinline__ float sigm_(float x) { return 1.0f / (1.0f + __expf(-x)); }

// ---------------- weight convert (f32 row-major [K][Nc] -> f16 transposed [Nc][Kpad]) ----
__global__ void convT_kernel(_Float16* __restrict__ dst, const float* __restrict__ src,
                             int K, int Nc, int Kpad) {
  long long i = (long long)blockIdx.x * blockDim.x + threadIdx.x;
  long long total = (long long)Nc * Kpad;
  if (i >= total) return;
  int k = (int)(i % Kpad);
  long long nc = i / Kpad;
  dst[i] = (k < K) ? (_Float16)src[(long long)k * Nc + nc] : (_Float16)0.0f;
}

__global__ void conv_f16_kernel(_Float16* __restrict__ dst, const float* __restrict__ src, int n) {
  int i = blockIdx.x * blockDim.x + threadIdx.x;
  if (i < n) dst[i] = (_Float16)src[i];
}

// ---------------- embedding gather: seqh[n][pos][b][0..255] = emb[caps[b][n][pos]] -------
__global__ void gather_kernel(_Float16* __restrict__ dst, const float* __restrict__ emb,
                              const int* __restrict__ caps) {
  int i = blockIdx.x * blockDim.x + threadIdx.x;     // N*T*B*EP = 204800
  if (i >= N_ * T_ * B_ * EP_) return;
  int e = i & (EP_ - 1);
  int r = i >> 8;               // (n*T+pos)*16 + b
  int b = r & 15; r >>= 4;
  int pos = r % T_;
  int n = r / T_;
  int idx = caps[(b * N_ + n) * T_ + pos];
  dst[i] = (e < E_) ? (_Float16)emb[(long long)idx * E_ + e] : (_Float16)0.0f;
}

// ---------------- generic WMMA GEMM, M-register-blocked -----------------------------------
// C[M][Nn] = A[M][K](f16) @ BT[Nn][K](f16, transposed) + bias
// Each wave: one 16-wide column tile x MREP row tiles; B fragment loaded once per k-chunk
// and reused for MREP WMMAs (cuts B-operand L2 traffic by MREP).
template <int MREP>
__global__ void gemm16_kernel(const _Float16* __restrict__ A, const _Float16* __restrict__ BT,
                              const float* __restrict__ bias,
                              float* __restrict__ Cf, _Float16* __restrict__ Ch,
                              int M, int Nn, int K, int relu) {
  int wave = (int)(((unsigned)blockIdx.x * blockDim.x + threadIdx.x) >> 5);
  int lane = threadIdx.x & 31;
  int Ntiles = Nn >> 4;
  int Mgrp = (M >> 4) / MREP;
  int total = Mgrp * Ntiles;
  if (wave >= total) return;
  int nt = wave % Ntiles, mg = wave / Ntiles;
  int hi = lane >> 4, cc = lane & 15;
  int col = nt * 16 + cc;
  float bv = bias ? bias[col] : 0.0f;
  F8 acc[MREP];
#pragma unroll
  for (int m = 0; m < MREP; m++)
#pragma unroll
    for (int i = 0; i < 8; i++) acc[m].f[i] = bv;

  const _Float16* arow0 = A + (size_t)(mg * MREP * 16 + cc) * K;  // A: M row = lane&15
  const _Float16* brow  = BT + (size_t)col * K;                   // B: N col = lane&15
  const size_t astr = (size_t)16 * K;
  for (int k0 = 0; k0 < K; k0 += 32) {
    H16 b;
    b.u[0] = *(const uint4*)(brow + k0 + 16 * hi);                // K {0..15|16..31}
    b.u[1] = *(const uint4*)(brow + k0 + 16 * hi + 8);
#pragma unroll
    for (int m = 0; m < MREP; m++) {
      H16 a;
      const _Float16* ar = arow0 + (size_t)m * astr;
      a.u[0] = *(const uint4*)(ar + k0 + 16 * hi);                // K runs {0..7|16..23}
      a.u[1] = *(const uint4*)(ar + k0 + 16 * hi + 8);            // K runs {8..15|24..31}
      acc[m].v = __builtin_amdgcn_wmma_f32_16x16x32_f16(false, a.v, false, b.v,
                                                        (short)0, acc[m].v, false, false);
    }
  }
#pragma unroll
  for (int m = 0; m < MREP; m++)
#pragma unroll
    for (int i = 0; i < 8; i++) {
      int row = (mg * MREP + m) * 16 + i + 8 * hi;
      float val = acc[m].f[i];
      if (relu) val = val > 0.0f ? val : 0.0f;
      if (Cf) Cf[(size_t)row * Nn + col] = val;
      if (Ch) Ch[(size_t)row * Nn + col] = (_Float16)val;
    }
}

// ---------------- one LSTM step, both directions; each wave = one 16-col H chunk ---------
// gates = xg + hprev(16xH f16) @ WhhT(4H x H f16);  c,h updated;  h emitted f16
__global__ void lstm_step_kernel(
    const _Float16* __restrict__ hprev_f, const _Float16* __restrict__ hprev_b,
    const _Float16* __restrict__ WT_f, const _Float16* __restrict__ WT_b,   // [2048][512]
    const float* __restrict__ xg_f, const float* __restrict__ xg_b,         // [16][2048] blocks
    float* __restrict__ cst_f, float* __restrict__ cst_b,                   // [16][512]
    _Float16* __restrict__ hnext_f, _Float16* __restrict__ hnext_b,         // [16][512]
    _Float16* __restrict__ out_f, _Float16* __restrict__ out_b,             // pre-offset
    int ldo) {
  int wave = (int)(((unsigned)blockIdx.x * blockDim.x + threadIdx.x) >> 5); // 0..63
  int lane = threadIdx.x & 31;
  int dir = wave >> 5;
  int chunk = wave & 31;                     // H chunk (32 chunks of 16)
  const _Float16* hp = dir ? hprev_b : hprev_f;
  const _Float16* WT = dir ? WT_b : WT_f;
  const float* xg    = dir ? xg_b : xg_f;
  float* cs          = dir ? cst_b : cst_f;
  _Float16* hn       = dir ? hnext_b : hnext_f;
  _Float16* op       = dir ? out_b : out_f;
  int hi = lane >> 4, cc = lane & 15;

  F8 acc[4];
#pragma unroll
  for (int g = 0; g < 4; g++) {
    int col = g * H_ + chunk * 16 + cc;
#pragma unroll
    for (int i = 0; i < 8; i++) acc[g].f[i] = xg[(size_t)(i + 8 * hi) * G4_ + col];
  }
  const _Float16* arow = hp + (size_t)cc * H_;
  for (int k0 = 0; k0 < H_; k0 += 32) {
    H16 a;
    a.u[0] = *(const uint4*)(arow + k0 + 16 * hi);
    a.u[1] = *(const uint4*)(arow + k0 + 16 * hi + 8);
#pragma unroll
    for (int g = 0; g < 4; g++) {
      H16 b;
      const _Float16* brow = WT + (size_t)(g * H_ + chunk * 16 + cc) * H_ + k0 + 16 * hi;
      b.u[0] = *(const uint4*)(brow);
      b.u[1] = *(const uint4*)(brow + 8);
      acc[g].v = __builtin_amdgcn_wmma_f32_16x16x32_f16(false, a.v, false, b.v,
                                                        (short)0, acc[g].v, false, false);
    }
  }
  int hcol = chunk * 16 + cc;
#pragma unroll
  for (int i = 0; i < 8; i++) {
    int b = i + 8 * hi;                      // batch row
    float ig = acc[0].f[i], fg = acc[1].f[i], gg = acc[2].f[i], og = acc[3].f[i];
    float cprev = cs[(size_t)b * H_ + hcol];
    float cv = sigm_(fg) * cprev + sigm_(ig) * tanhf(gg);
    float hv = sigm_(og) * tanhf(cv);
    cs[(size_t)b * H_ + hcol] = cv;
    hn[(size_t)b * H_ + hcol] = (_Float16)hv;
    op[(size_t)b * ldo + hcol] = (_Float16)hv;
  }
}

// ---------------- broadcast initial MLP states into 4 slots ------------------------------
__global__ void init_state_kernel(const float* __restrict__ h0, const float* __restrict__ c0,
                                  float* __restrict__ cstate, _Float16* __restrict__ hbuf) {
  int i = blockIdx.x * blockDim.x + threadIdx.x;
  if (i >= B_ * H_) return;
  float hv = h0[i], cv = c0[i];
  _Float16 hh = (_Float16)hv;
#pragma unroll
  for (int s = 0; s < 4; s++) {
    cstate[s * B_ * H_ + i] = cv;
    hbuf[(s * 2 + 0) * B_ * H_ + i] = hh;    // parity 0
  }
}

// ================================ host orchestration =====================================
extern "C" void kernel_launch(void* const* d_in, const int* in_sizes, int n_in,
                              void* d_out, int out_size, void* d_ws, size_t ws_size,
                              hipStream_t stream) {
  (void)in_sizes; (void)n_in; (void)out_size; (void)ws_size;
  const float* img  = (const float*)d_in[0];
  const int*   caps = (const int*)d_in[1];
  const float* Wh1 = (const float*)d_in[2],  *bh1 = (const float*)d_in[3];
  const float* Wh2 = (const float*)d_in[4],  *bh2 = (const float*)d_in[5];
  const float* Wc1 = (const float*)d_in[6],  *bc1 = (const float*)d_in[7];
  const float* Wc2 = (const float*)d_in[8],  *bc2 = (const float*)d_in[9];
  const float* emb = (const float*)d_in[10];
  const float* Wih0f = (const float*)d_in[11], *Whh0f = (const float*)d_in[12], *b0f = (const float*)d_in[13];
  const float* Wih0b = (const float*)d_in[14], *Whh0b = (const float*)d_in[15], *b0b = (const float*)d_in[16];
  const float* Wih1f = (const float*)d_in[17], *Whh1f = (const float*)d_in[18], *b1f = (const float*)d_in[19];
  const float* Wih1b = (const float*)d_in[20], *Whh1b = (const float*)d_in[21], *b1b = (const float*)d_in[22];
  const float* Wfc = (const float*)d_in[23], *bfc = (const float*)d_in[24];
  float* out = (float*)d_out;

  char* w = (char*)d_ws;
  size_t off = 0;
  auto alloc = [&](size_t bytes) -> void* {
    off = (off + 255) & ~(size_t)255;
    void* p = w + off; off += bytes; return p;
  };
  // f16 transposed weights
  _Float16* Wh1T   = (_Float16*)alloc((size_t)H2_ * F_ * 2);   // [1024][2048]
  _Float16* Wh2T   = (_Float16*)alloc((size_t)H_ * H2_ * 2);   // [512][1024]
  _Float16* Wc1T   = (_Float16*)alloc((size_t)H2_ * F_ * 2);
  _Float16* Wc2T   = (_Float16*)alloc((size_t)H_ * H2_ * 2);
  _Float16* Wih0fT = (_Float16*)alloc((size_t)G4_ * EP_ * 2);  // [2048][256] (K padded 250->256)
  _Float16* Wih0bT = (_Float16*)alloc((size_t)G4_ * EP_ * 2);
  _Float16* Whh0fT = (_Float16*)alloc((size_t)G4_ * H_ * 2);   // [2048][512]
  _Float16* Whh0bT = (_Float16*)alloc((size_t)G4_ * H_ * 2);
  _Float16* Wih1fT = (_Float16*)alloc((size_t)G4_ * H2_ * 2);  // [2048][1024]
  _Float16* Wih1bT = (_Float16*)alloc((size_t)G4_ * H2_ * 2);
  _Float16* Whh1fT = (_Float16*)alloc((size_t)G4_ * H_ * 2);
  _Float16* Whh1bT = (_Float16*)alloc((size_t)G4_ * H_ * 2);
  _Float16* WfcT   = (_Float16*)alloc((size_t)V_ * H2_ * 2);   // [30000][1024]
  // activations
  _Float16* imgh = (_Float16*)alloc((size_t)B_ * F_ * 2);
  _Float16* t1h  = (_Float16*)alloc((size_t)B_ * H2_ * 2);
  _Float16* t2h  = (_Float16*)alloc((size_t)B_ * H2_ * 2);
  float*    h0f  = (float*)alloc((size_t)B_ * H_ * 4);
  float*    c0f  = (float*)alloc((size_t)B_ * H_ * 4);
  _Float16* seqh = (_Float16*)alloc((size_t)N_ * T_ * B_ * EP_ * 2);
  float*    xg0f = (float*)alloc((size_t)N_ * T_ * B_ * G4_ * 4);
  float*    xg0b = (float*)alloc((size_t)N_ * T_ * B_ * G4_ * 4);
  _Float16* x1   = (_Float16*)alloc((size_t)T_ * B_ * H2_ * 2);
  float*    xg1f = (float*)alloc((size_t)T_ * B_ * G4_ * 4);
  float*    xg1b = (float*)alloc((size_t)T_ * B_ * G4_ * 4);
  _Float16* outh = (_Float16*)alloc((size_t)N_ * T_ * B_ * H2_ * 2);
  float*    cstate = (float*)alloc((size_t)4 * B_ * H_ * 4);
  _Float16* hbuf   = (_Float16*)alloc((size_t)8 * B_ * H_ * 2); // 4 slots x 2 parities

  auto convT = [&](_Float16* dst, const float* src, int K, int Nc, int Kpad) {
    long long tot = (long long)Nc * Kpad;
    int blocks = (int)((tot + 255) / 256);
    convT_kernel<<<blocks, 256, 0, stream>>>(dst, src, K, Nc, Kpad);
  };
  auto gemm = [&](const _Float16* A, const _Float16* BT, const float* bias,
                  float* Cf, _Float16* Ch, int M, int Nn, int K, int relu) {
    int Mtiles = M / 16, Ntiles = Nn / 16;
    int mrep = (Mtiles % 5 == 0) ? 5 : (Mtiles % 2 == 0) ? 2 : 1;
    int tiles = (Mtiles / mrep) * Ntiles;
    int blocks = (tiles + 7) / 8;
    if (mrep == 5)
      gemm16_kernel<5><<<blocks, 256, 0, stream>>>(A, BT, bias, Cf, Ch, M, Nn, K, relu);
    else if (mrep == 2)
      gemm16_kernel<2><<<blocks, 256, 0, stream>>>(A, BT, bias, Cf, Ch, M, Nn, K, relu);
    else
      gemm16_kernel<1><<<blocks, 256, 0, stream>>>(A, BT, bias, Cf, Ch, M, Nn, K, relu);
  };

  // 1) weight conversion (f32 -> transposed f16)
  convT(Wh1T,   Wh1,   F_,  H2_, F_);
  convT(Wh2T,   Wh2,   H2_, H_,  H2_);
  convT(Wc1T,   Wc1,   F_,  H2_, F_);
  convT(Wc2T,   Wc2,   H2_, H_,  H2_);
  convT(Wih0fT, Wih0f, E_,  G4_, EP_);
  convT(Wih0bT, Wih0b, E_,  G4_, EP_);
  convT(Whh0fT, Whh0f, H_,  G4_, H_);
  convT(Whh0bT, Whh0b, H_,  G4_, H_);
  convT(Wih1fT, Wih1f, H2_, G4_, H2_);
  convT(Wih1bT, Wih1b, H2_, G4_, H2_);
  convT(Whh1fT, Whh1f, H_,  G4_, H_);
  convT(Whh1bT, Whh1b, H_,  G4_, H_);
  convT(WfcT,   Wfc,   H2_, V_,  H2_);
  conv_f16_kernel<<<(B_ * F_ + 255) / 256, 256, 0, stream>>>(imgh, img, B_ * F_);

  // 2) MLP state init: h = relu(relu(img@Wh1+bh1)@Wh2+bh2), c likewise
  gemm(imgh, Wh1T, bh1, nullptr, t1h, B_, H2_, F_, 1);
  gemm(t1h,  Wh2T, bh2, h0f, nullptr, B_, H_, H2_, 1);
  gemm(imgh, Wc1T, bc1, nullptr, t2h, B_, H2_, F_, 1);
  gemm(t2h,  Wc2T, bc2, c0f, nullptr, B_, H_, H2_, 1);
  init_state_kernel<<<(B_ * H_ + 255) / 256, 256, 0, stream>>>(h0f, c0f, cstate, hbuf);

  // 3) embedding gather + hoisted layer-0 input-gate preactivations (once for all prefixes)
  gather_kernel<<<(N_ * T_ * B_ * EP_ + 255) / 256, 256, 0, stream>>>(seqh, emb, caps);
  gemm(seqh, Wih0fT, b0f, xg0f, nullptr, N_ * T_ * B_, G4_, EP_, 0);
  gemm(seqh, Wih0bT, b0b, xg0b, nullptr, N_ * T_ * B_, G4_, EP_, 0);

  // 4) sequential bilstm2 chain: 50 calls, states threaded through all of them
  int p[4] = {0, 0, 0, 0};
  auto hb = [&](int slot, int par) -> _Float16* { return hbuf + (size_t)(slot * 2 + par) * B_ * H_; };
  auto cs = [&](int slot) -> float* { return cstate + (size_t)slot * B_ * H_; };

  for (int k = 0; k < 50; k++) {
    int t = k / N_, n = k % N_, len = t + 1;
    // layer 0 scan (fwd + bwd per step), writes f16 into x1 concat buffer
    for (int s = 0; s < len; s++) {
      int posf = s, posb = len - 1 - s;
      lstm_step_kernel<<<8, 256, 0, stream>>>(
          hb(0, p[0]), hb(1, p[1]), Whh0fT, Whh0bT,
          xg0f + (size_t)((n * T_ + posf) * B_) * G4_,
          xg0b + (size_t)((n * T_ + posb) * B_) * G4_,
          cs(0), cs(1), hb(0, p[0] ^ 1), hb(1, p[1] ^ 1),
          x1 + (size_t)posf * B_ * H2_,
          x1 + (size_t)posb * B_ * H2_ + H_, H2_);
      p[0] ^= 1; p[1] ^= 1;
    }
    // layer 1 input gates for this prefix
    gemm(x1, Wih1fT, b1f, xg1f, nullptr, len * B_, G4_, H2_, 0);
    gemm(x1, Wih1bT, b1b, xg1b, nullptr, len * B_, G4_, H2_, 0);
    // layer 1 scan, writes f16 into outh (final t=9 writes win deterministically)
    for (int s = 0; s < len; s++) {
      int posf = s, posb = len - 1 - s;
      lstm_step_kernel<<<8, 256, 0, stream>>>(
          hb(2, p[2]), hb(3, p[3]), Whh1fT, Whh1bT,
          xg1f + (size_t)(posf * B_) * G4_,
          xg1b + (size_t)(posb * B_) * G4_,
          cs(2), cs(3), hb(2, p[2] ^ 1), hb(3, p[3] ^ 1),
          outh + (size_t)((n * T_ + posf) * B_) * H2_,
          outh + (size_t)((n * T_ + posb) * B_) * H2_ + H_, H2_);
      p[2] ^= 1; p[3] ^= 1;
    }
  }

  // 5) final FC: (N*T*B=800, 1024) @ (1024, 30000) + bfc -> d_out (f32)
  gemm(outh, WfcT, bfc, out, nullptr, N_ * T_ * B_, V_, H2_, 0);
}
